// PointNetSetAbstraction_44444321579356
// MI455X (gfx1250) — compile-verified, matrix operand order checked by reading, not run on hardware
//
#include <hip/hip_runtime.h>
#include <hip/hip_bf16.h>

typedef __attribute__((ext_vector_type(8)))  float  v8f;
typedef __attribute__((ext_vector_type(16))) __bf16 v16bf;
typedef __attribute__((ext_vector_type(8)))  __bf16 v8bf;

#define B_   8
#define N_   4096
#define S_   1024
#define NS_  32
#define M_   (B_ * S_ * NS_)   /* 262144 rows */
#define MT_  (M_ / 16)         /* 16384 M-tiles */
#define RAD2 0.04f

__device__ __forceinline__ unsigned short f2bf(float x) {
  unsigned u = __float_as_uint(x);
  unsigned r = ((u >> 16) & 1u) + 0x7FFFu;   // round-to-nearest-even
  return (unsigned short)((u + r) >> 16);
}

// ---------------------------------------------------------------------------
// 1) Farthest point sampling: one workgroup per batch, xyz in LDS,
//    per-thread distance state in registers, wave32 shuffle reductions.
// ---------------------------------------------------------------------------
__global__ __launch_bounds__(512) void fps_kernel(const float* __restrict__ xyz,
                                                  float* __restrict__ newxyz) {
  __shared__ float X[N_], Y[N_], Z[N_];
  __shared__ int   FI[S_];
  __shared__ float wv[16];
  __shared__ int   wi[16];
  __shared__ int   winner;

  const int b = blockIdx.x, tid = threadIdx.x;
  const int lane = tid & 31, wave = tid >> 5;
  const float* base = xyz + (size_t)b * N_ * 3;
  for (int j = tid; j < N_; j += 512) {
    X[j] = base[j * 3 + 0];
    Y[j] = base[j * 3 + 1];
    Z[j] = base[j * 3 + 2];
  }
  float dist[8];
#pragma unroll
  for (int k = 0; k < 8; ++k) dist[k] = 1e10f;
  __syncthreads();

  int far = 0;
  for (int it = 0; it < S_; ++it) {
    if (tid == 0) FI[it] = far;
    const float cx = X[far], cy = Y[far], cz = Z[far];
    float bv = -1.0f;
    int   bi = 0x7fffffff;
#pragma unroll
    for (int k = 0; k < 8; ++k) {
      const int j = tid + k * 512;
      const float dx = X[j] - cx, dy = Y[j] - cy, dz = Z[j] - cz;
      const float d  = dx * dx + dy * dy + dz * dz;
      const float nd = fminf(dist[k], d);
      dist[k] = nd;
      if (nd > bv || (nd == bv && j < bi)) { bv = nd; bi = j; }
    }
#pragma unroll
    for (int m = 1; m < 32; m <<= 1) {
      const float ov = __shfl_xor(bv, m, 32);
      const int   oi = __shfl_xor(bi, m, 32);
      if (ov > bv || (ov == bv && oi < bi)) { bv = ov; bi = oi; }
    }
    if (lane == 0) { wv[wave] = bv; wi[wave] = bi; }
    __syncthreads();
    if (wave == 0) {
      float v2 = (lane < 16) ? wv[lane] : -1.0f;
      int   i2 = (lane < 16) ? wi[lane] : 0x7fffffff;
#pragma unroll
      for (int m = 1; m < 16; m <<= 1) {
        const float ov = __shfl_xor(v2, m, 32);
        const int   oi = __shfl_xor(i2, m, 32);
        if (ov > v2 || (ov == v2 && oi < i2)) { v2 = ov; i2 = oi; }
      }
      if (lane == 0) winner = i2;
    }
    __syncthreads();
    far = winner;
  }
  // new_xyz -> first B*S*3 floats of d_out
  for (int s = tid; s < S_; s += 512) {
    const int k = FI[s];
    newxyz[(size_t)(b * S_ + s) * 3 + 0] = X[k];
    newxyz[(size_t)(b * S_ + s) * 3 + 1] = Y[k];
    newxyz[(size_t)(b * S_ + s) * 3 + 2] = Z[k];
  }
}

// ---------------------------------------------------------------------------
// 2) Ordered ball query + grouping: one wave per (b,s); the lane holding a
//    selected point writes the full 96-channel bf16 row (row-major A matrix).
// ---------------------------------------------------------------------------
__device__ __forceinline__ void write_row(unsigned* rp, const float* xb, const float* pb,
                                          int j, float cx, float cy, float cz) {
  const float a0 = xb[j * 3 + 0] - cx;
  const float a1 = xb[j * 3 + 1] - cy;
  const float a2 = xb[j * 3 + 2] - cz;
  const float* p = pb + (size_t)j * 64;
  rp[0] = (unsigned)f2bf(a0) | ((unsigned)f2bf(a1) << 16);
  rp[1] = (unsigned)f2bf(a2) | ((unsigned)f2bf(p[0]) << 16);
#pragma unroll
  for (int k = 2; k <= 32; ++k)
    rp[k] = (unsigned)f2bf(p[2 * k - 3]) | ((unsigned)f2bf(p[2 * k - 2]) << 16);
  rp[33] = (unsigned)f2bf(p[63]);   // ch66 = p[63], ch67 = 0
#pragma unroll
  for (int k = 34; k < 48; ++k) rp[k] = 0u;   // K padding 68..95
}

__global__ __launch_bounds__(256) void group_kernel(const float* __restrict__ xyz,
                                                    const float* __restrict__ pts,
                                                    const float* __restrict__ newxyz,
                                                    unsigned short* __restrict__ H0) {
  const int tid = threadIdx.x, lane = tid & 31;
  const int g = blockIdx.x * 8 + (tid >> 5);   // b*S + s
  const int b = g >> 10;
  const float cx = newxyz[g * 3 + 0];
  const float cy = newxyz[g * 3 + 1];
  const float cz = newxyz[g * 3 + 2];
  const float* xb = xyz + (size_t)b * N_ * 3;
  const float* pb = pts + (size_t)b * N_ * 64;

  int found = 0, j0 = -1;
  for (int base0 = 0; base0 < N_ && found < NS_; base0 += 32) {
    const int j = base0 + lane;
    const float dx = xb[j * 3 + 0] - cx;
    const float dy = xb[j * 3 + 1] - cy;
    const float dz = xb[j * 3 + 2] - cz;
    const bool pred = (dx * dx + dy * dy + dz * dz) <= RAD2;
    const unsigned mask = (unsigned)__ballot(pred);
    if (j0 < 0 && mask) j0 = base0 + (__ffs(mask) - 1);
    const int pos = found + __popc(mask & ((1u << lane) - 1u));
    if (pred && pos < NS_) {
      unsigned* rp = (unsigned*)(H0 + ((size_t)g * NS_ + pos) * 96);
      write_row(rp, xb, pb, j, cx, cy, cz);
    }
    found += __popc(mask);
  }
  if (found < NS_) {            // pad with first index (reference semantics)
    if (j0 < 0) j0 = 0;
    const int pad = NS_ - found;
    if (lane < pad) {
      unsigned* rp = (unsigned*)(H0 + ((size_t)g * NS_ + found + lane) * 96);
      write_row(rp, xb, pb, j0, cx, cy, cz);
    }
  }
}

// ---------------------------------------------------------------------------
// 3) Weight packing: row-major [Cout][Kpad] bf16 (zero pad K 67->96 for L0).
// ---------------------------------------------------------------------------
__global__ void pack_weights(const float* __restrict__ w0, const float* __restrict__ w1,
                             const float* __restrict__ w2, unsigned short* __restrict__ Wb0,
                             unsigned short* __restrict__ Wb1, unsigned short* __restrict__ Wb2) {
  const int i = blockIdx.x * blockDim.x + threadIdx.x;
  const int stride = gridDim.x * blockDim.x;
  for (int idx = i; idx < 64 * 96; idx += stride) {
    const int n = idx / 96, k = idx % 96;
    Wb0[idx] = (k < 67) ? f2bf(w0[n * 67 + k]) : (unsigned short)0;
  }
  for (int idx = i; idx < 64 * 64; idx += stride) Wb1[idx] = f2bf(w1[idx]);
  for (int idx = i; idx < 128 * 64; idx += stride) Wb2[idx] = f2bf(w2[idx]);
}

// ---------------------------------------------------------------------------
// WMMA GEMM core: one wave owns one 16-row M-tile, all NT column tiles.
// Row-major bf16 A/W feed fragments with two 16B loads per lane each
// (matches the documented 16-bit A/B fragment striping).
// ---------------------------------------------------------------------------
template <int NT>
__device__ __forceinline__ void zero_acc(v8f acc[NT]) {
#pragma unroll
  for (int nt = 0; nt < NT; ++nt) {
#pragma unroll
    for (int i = 0; i < 8; ++i) acc[nt][i] = 0.0f;
  }
}

template <int KT, int NT>
__device__ __forceinline__ void gemm_tile(const unsigned short* __restrict__ A,
                                          const unsigned short* __restrict__ W,
                                          int tile, int lane, v8f acc[NT]) {
  const int Kpad = KT * 32;
  const int rsel = lane & 15;
  const int hsel = (lane >> 4) * 8;
  const unsigned short* arow = A + ((size_t)tile * 16 + rsel) * Kpad + hsel;
#pragma unroll
  for (int kt = 0; kt < KT; ++kt) {
    const unsigned short* ap = arow + kt * 32;
    const v8bf alo = *(const v8bf*)ap;
    const v8bf ahi = *(const v8bf*)(ap + 16);
    const v16bf a = __builtin_shufflevector(alo, ahi, 0, 1, 2, 3, 4, 5, 6, 7,
                                            8, 9, 10, 11, 12, 13, 14, 15);
#pragma unroll
    for (int nt = 0; nt < NT; ++nt) {
      const unsigned short* bp = W + ((size_t)(nt * 16 + rsel)) * Kpad + kt * 32 + hsel;
      const v8bf blo = *(const v8bf*)bp;
      const v8bf bhi = *(const v8bf*)(bp + 16);
      const v16bf bb = __builtin_shufflevector(blo, bhi, 0, 1, 2, 3, 4, 5, 6, 7,
                                               8, 9, 10, 11, 12, 13, 14, 15);
      acc[nt] = __builtin_amdgcn_wmma_f32_16x16x32_bf16(false, a, false, bb,
                                                        (short)0, acc[nt], false, false);
    }
  }
}

// 4a) GEMM + per-tile channel sums (deterministic partials, no atomics).
template <int KT, int NT>
__global__ __launch_bounds__(256) void gemm_stats(const unsigned short* __restrict__ A,
                                                  const unsigned short* __restrict__ W,
                                                  float* __restrict__ pSum,
                                                  float* __restrict__ pSq) {
  const int tid = threadIdx.x, lane = tid & 31, wave = tid >> 5;
  const int tile = blockIdx.x * 8 + wave;
  v8f acc[NT];
  zero_acc<NT>(acc);
  gemm_tile<KT, NT>(A, W, tile, lane, acc);
  const int C = NT * 16;
#pragma unroll
  for (int nt = 0; nt < NT; ++nt) {
    float s1 = 0.0f, s2 = 0.0f;
#pragma unroll
    for (int v = 0; v < 8; ++v) {
      const float z = acc[nt][v];
      s1 += z;
      s2 += z * z;
    }
    s1 += __shfl_xor(s1, 16, 32);
    s2 += __shfl_xor(s2, 16, 32);
    if (lane < 16) {
      pSum[(size_t)tile * C + nt * 16 + lane] = s1;
      pSq[(size_t)tile * C + nt * 16 + lane]  = s2;
    }
  }
}

// 4b) Fixed-order tree reduction of the 16384 tile partials per channel.
__global__ __launch_bounds__(256) void reduce_stats(const float* __restrict__ pSum,
                                                    const float* __restrict__ pSq,
                                                    float* __restrict__ sum,
                                                    float* __restrict__ sq, int C) {
  __shared__ float ls[256], lq[256];
  const int c = blockIdx.x, t = threadIdx.x;
  float s = 0.0f, q = 0.0f;
  for (int i = t; i < MT_; i += 256) {
    s += pSum[(size_t)i * C + c];
    q += pSq[(size_t)i * C + c];
  }
  ls[t] = s;
  lq[t] = q;
  __syncthreads();
  for (int st = 128; st > 0; st >>= 1) {
    if (t < st) { ls[t] += ls[t + st]; lq[t] += lq[t + st]; }
    __syncthreads();
  }
  if (t == 0) { sum[c] = ls[0]; sq[c] = lq[0]; }
}

// 4c) Fold batch stats + gamma/beta into per-channel scale/shift.
//     (Bias b cancels exactly under batchnorm, so it is omitted.)
__global__ void bn_finalize(const float* __restrict__ sum, const float* __restrict__ sq,
                            const float* __restrict__ g, const float* __restrict__ be,
                            float* __restrict__ scale, float* __restrict__ shift, int C) {
  const int c = threadIdx.x;
  if (c < C) {
    const float invCnt = 1.0f / (float)M_;
    const float mu  = sum[c] * invCnt;
    const float var = sq[c] * invCnt - mu * mu;
    const float sc  = g[c] * rsqrtf(var + 1e-5f);
    scale[c] = sc;
    shift[c] = be[c] - mu * sc;
  }
}

// 4d) Recompute GEMM fused with BN+ReLU, emit next layer's bf16 A-matrix.
template <int KT, int NT>
__global__ __launch_bounds__(256) void gemm_apply_hidden(const unsigned short* __restrict__ A,
                                                         const unsigned short* __restrict__ W,
                                                         const float* __restrict__ scale,
                                                         const float* __restrict__ shift,
                                                         unsigned short* __restrict__ Hout) {
  const int tid = threadIdx.x, lane = tid & 31, wave = tid >> 5;
  const int tile = blockIdx.x * 8 + wave;
  v8f acc[NT];
  zero_acc<NT>(acc);
  gemm_tile<KT, NT>(A, W, tile, lane, acc);
  const int C = NT * 16;           // == next layer Kpad (64)
  const int n0 = lane & 15;
  const int mhi = (lane >> 4) * 8;
#pragma unroll
  for (int nt = 0; nt < NT; ++nt) {
    const int c = nt * 16 + n0;
    const float sc = scale[c], sh = shift[c];
#pragma unroll
    for (int v = 0; v < 8; ++v) {
      const int row = tile * 16 + mhi + v;
      const float y = fmaxf(acc[nt][v] * sc + sh, 0.0f);
      Hout[(size_t)row * C + c] = f2bf(y);
    }
  }
}

// 4e) Last layer: GEMM + BN + ReLU + max over 32 samples + transposed store.
template <int KT, int NT>
__global__ __launch_bounds__(256) void gemm_apply_final(const unsigned short* __restrict__ A,
                                                        const unsigned short* __restrict__ W,
                                                        const float* __restrict__ scale,
                                                        const float* __restrict__ shift,
                                                        float* __restrict__ outp) {
  __shared__ float wmax[8][NT * 16];
  const int tid = threadIdx.x, lane = tid & 31, wave = tid >> 5;
  const int tile = blockIdx.x * 8 + wave;
  v8f acc[NT];
  zero_acc<NT>(acc);
  gemm_tile<KT, NT>(A, W, tile, lane, acc);
  const int C = NT * 16;   // 128
  const int n0 = lane & 15;
#pragma unroll
  for (int nt = 0; nt < NT; ++nt) {
    const int c = nt * 16 + n0;
    const float sc = scale[c], sh = shift[c];
    float mx = -1e30f;
#pragma unroll
    for (int v = 0; v < 8; ++v) {
      const float y = fmaxf(acc[nt][v] * sc + sh, 0.0f);
      mx = fmaxf(mx, y);
    }
    mx = fmaxf(mx, __shfl_xor(mx, 16, 32));
    if (lane < 16) wmax[wave][c] = mx;
  }
  __syncthreads();
  // 8 tiles per block = 4 (b,s) groups of 32 samples; combine tile pairs.
  for (int idx = tid; idx < 4 * C; idx += 256) {
    const int q = idx / C, c = idx % C;
    const float m = fmaxf(wmax[2 * q][c], wmax[2 * q + 1][c]);
    const int g = blockIdx.x * 4 + q;     // b*S + s
    const int b = g >> 10, s = g & 1023;
    outp[(size_t)(b * 128 + c) * 1024 + s] = m;
  }
}

// ---------------------------------------------------------------------------
extern "C" void kernel_launch(void* const* d_in, const int* in_sizes, int n_in,
                              void* d_out, int out_size, void* d_ws, size_t ws_size,
                              hipStream_t stream) {
  (void)in_sizes; (void)n_in; (void)out_size; (void)ws_size;
  const float* xyz = (const float*)d_in[0];
  const float* pts = (const float*)d_in[1];
  const float* w0  = (const float*)d_in[2];
  const float* g0  = (const float*)d_in[4];
  const float* be0 = (const float*)d_in[5];
  const float* w1  = (const float*)d_in[6];
  const float* g1  = (const float*)d_in[8];
  const float* be1 = (const float*)d_in[9];
  const float* w2  = (const float*)d_in[10];
  const float* g2  = (const float*)d_in[12];
  const float* be2 = (const float*)d_in[13];
  float* out = (float*)d_out;

  // Workspace partition (256B aligned).
  char* wsp = (char*)d_ws;
  auto take = [&](size_t bytes) -> char* {
    char* p = wsp;
    wsp += (bytes + 255) & ~(size_t)255;
    return p;
  };
  unsigned short* Wb0 = (unsigned short*)take((size_t)64 * 96 * 2);
  unsigned short* Wb1 = (unsigned short*)take((size_t)64 * 64 * 2);
  unsigned short* Wb2 = (unsigned short*)take((size_t)128 * 64 * 2);
  float* sumA = (float*)take(128 * 4);
  float* sqA  = (float*)take(128 * 4);
  float* scl  = (float*)take(128 * 4);
  float* shf  = (float*)take(128 * 4);
  float* pSum = (float*)take((size_t)MT_ * 128 * 4);
  float* pSq  = (float*)take((size_t)MT_ * 128 * 4);
  unsigned short* H0 = (unsigned short*)take((size_t)M_ * 96 * 2);
  unsigned short* H1 = (unsigned short*)take((size_t)M_ * 64 * 2);
  unsigned short* H2 = (unsigned short*)take((size_t)M_ * 64 * 2);

  float* newxyz = out;                       // first B*S*3 floats
  float* outfeat = out + (size_t)B_ * S_ * 3;

  pack_weights<<<80, 256, 0, stream>>>(w0, w1, w2, Wb0, Wb1, Wb2);
  fps_kernel<<<B_, 512, 0, stream>>>(xyz, newxyz);
  group_kernel<<<(B_ * S_) / 8, 256, 0, stream>>>(xyz, pts, newxyz, H0);

  // Layer 0: K=96, Cout=64
  gemm_stats<3, 4><<<MT_ / 8, 256, 0, stream>>>(H0, Wb0, pSum, pSq);
  reduce_stats<<<64, 256, 0, stream>>>(pSum, pSq, sumA, sqA, 64);
  bn_finalize<<<1, 128, 0, stream>>>(sumA, sqA, g0, be0, scl, shf, 64);
  gemm_apply_hidden<3, 4><<<MT_ / 8, 256, 0, stream>>>(H0, Wb0, scl, shf, H1);

  // Layer 1: K=64, Cout=64
  gemm_stats<2, 4><<<MT_ / 8, 256, 0, stream>>>(H1, Wb1, pSum, pSq);
  reduce_stats<<<64, 256, 0, stream>>>(pSum, pSq, sumA, sqA, 64);
  bn_finalize<<<1, 128, 0, stream>>>(sumA, sqA, g1, be1, scl, shf, 64);
  gemm_apply_hidden<2, 4><<<MT_ / 8, 256, 0, stream>>>(H1, Wb1, scl, shf, H2);

  // Layer 2: K=64, Cout=128, fused max-pool + transpose into d_out
  gemm_stats<2, 8><<<MT_ / 8, 256, 0, stream>>>(H2, Wb2, pSum, pSq);
  reduce_stats<<<128, 256, 0, stream>>>(pSum, pSq, sumA, sqA, 128);
  bn_finalize<<<1, 128, 0, stream>>>(sumA, sqA, g2, be2, scl, shf, 128);
  gemm_apply_final<2, 8><<<MT_ / 8, 256, 0, stream>>>(H2, Wb2, scl, shf, outfeat);
}
